// MultiLevelGraphTokenizer_9509057593391
// MI455X (gfx1250) — compile-verified, matrix operand order checked by reading, not run on hardware
//
#include <hip/hip_runtime.h>
#include <hip/hip_bf16.h>

#define NN 10000
#define NE 320000
#define HD 256
#define NWORDS 313   // ceil(10000/32)

typedef __attribute__((ext_vector_type(16))) _Float16 v16h;
typedef __attribute__((ext_vector_type(8)))  float    v8f;
typedef __attribute__((ext_vector_type(4)))  float    v4f;

__device__ __forceinline__ int wmma_k_of(int i, int lane) {
    // 16-bit A/B fragment K index for half-slot i in lane (CDNA5 ISA 7.12.2)
    return (i & 7) + 8 * (2 * (i >> 3) + (lane >> 4));
}

// Repack f32 weight matrix [ksteps*32, 256] into fragment-major f16:
// P[((kstep*16 + ntile)*32 + lane)*16 + i] = W[(kstep*32 + k_of(i,lane))*256 + ntile*16 + lane%16]
__global__ __launch_bounds__(32) void pack_w(const float* __restrict__ W, _Float16* __restrict__ P) {
    const int t = blockIdx.x;           // tile = kstep*16 + ntile
    const int lane = threadIdx.x;
    const int ks = t >> 4, nt = t & 15;
    v16h tmp;
    #pragma unroll
    for (int i = 0; i < 16; ++i) {
        int k = ks * 32 + wmma_k_of(i, lane);
        tmp[i] = (_Float16)W[(size_t)k * HD + nt * 16 + (lane & 15)];
    }
    *(v16h*)(P + ((size_t)t * 32 + lane) * 16) = tmp;
}

// C[M,256] = act( [A0|A1|A2] @ W + bias ), Wp = fragment-major packed f16 (parts*8 ksteps)
__global__ __launch_bounds__(256) void gemm_wmma(
    const float* __restrict__ A0, const float* __restrict__ A1, const float* __restrict__ A2,
    const _Float16* __restrict__ Wp, const float* __restrict__ bias,
    float* __restrict__ C, int M, int parts, int do_relu)
{
    __shared__ _Float16 As[16][34];           // 16x32 A tile (padded)
    const int tid  = threadIdx.x;
    const int lane = tid & 31;
    const int wave = tid >> 5;                // 8 waves, 32 cols each
    const int m0   = blockIdx.x * 16;
    const int n0   = wave * 32;

    v8f acc0 = {}; v8f acc1 = {};

    const int ksteps = parts * 8;             // 32-wide K steps
    for (int ks = 0; ks < ksteps; ++ks) {
        const int part = ks >> 3;
        const int kl   = (ks & 7) * 32;
        const float* Ap = (part == 0) ? A0 : ((part == 1) ? A1 : A2);

        __syncthreads();
        #pragma unroll
        for (int e = 0; e < 2; ++e) {
            int idx = tid + e * 256;
            int r = idx >> 5, c = idx & 31;
            As[r][c] = (_Float16)Ap[(size_t)(m0 + r) * HD + kl + c];
        }
        __syncthreads();

        v16h a;
        const int m = lane & 15;
        #pragma unroll
        for (int i = 0; i < 16; ++i) a[i] = As[m][wmma_k_of(i, lane)];

        // packed B fragments: one 32B contiguous load per tile per lane
        const v16h b0 = *(const v16h*)(Wp + (((size_t)ks * 16 + wave * 2)     * 32 + lane) * 16);
        const v16h b1 = *(const v16h*)(Wp + (((size_t)ks * 16 + wave * 2 + 1) * 32 + lane) * 16);

        acc0 = __builtin_amdgcn_wmma_f32_16x16x32_f16(false, a, false, b0, (short)0, acc0, false, false);
        acc1 = __builtin_amdgcn_wmma_f32_16x16x32_f16(false, a, false, b1, (short)0, acc1, false, false);
    }

    const int n = lane & 15;
    const int mrow = 8 * (lane >> 4);
    #pragma unroll
    for (int r = 0; r < 8; ++r) {
        int row = m0 + r + mrow;
        float v0 = acc0[r], v1 = acc1[r];
        if (bias) { v0 += bias[n0 + n]; v1 += bias[n0 + 16 + n]; }
        if (do_relu) { v0 = fmaxf(v0, 0.f); v1 = fmaxf(v1, 0.f); }
        C[(size_t)row * HD + n0 + n]      = v0;
        C[(size_t)row * HD + n0 + 16 + n] = v1;
    }
}

__global__ void count_deg(const int* __restrict__ row, int* __restrict__ cnt, int e) {
    int i = blockIdx.x * blockDim.x + threadIdx.x;
    if (i < e) atomicAdd(&cnt[row[i]], 1);
}

__global__ __launch_bounds__(256) void scan_k(const int* __restrict__ cnt,
                                              int* __restrict__ row_ptr,
                                              int* __restrict__ cursor, int n, int etot) {
    __shared__ int buf[256];
    __shared__ int carry;
    int tid = threadIdx.x;
    if (tid == 0) carry = 0;
    __syncthreads();
    for (int base = 0; base < n; base += 256) {
        int v = (base + tid < n) ? cnt[base + tid] : 0;
        buf[tid] = v;
        __syncthreads();
        #pragma unroll
        for (int off = 1; off < 256; off <<= 1) {
            int t = (tid >= off) ? buf[tid - off] : 0;
            __syncthreads();
            buf[tid] += t;
            __syncthreads();
        }
        int excl = buf[tid] - v;
        if (base + tid < n) { row_ptr[base + tid] = carry + excl; cursor[base + tid] = carry + excl; }
        __syncthreads();
        if (tid == 0) carry += buf[255];
        __syncthreads();
    }
    if (tid == 0) row_ptr[n] = etot;
}

__global__ void scatter_edges(const int* __restrict__ row, const int* __restrict__ col,
                              int* __restrict__ cursor, int* __restrict__ col_idx, int e) {
    int i = blockIdx.x * blockDim.x + threadIdx.x;
    if (i < e) {
        int p = atomicAdd(&cursor[row[i]], 1);
        col_idx[p] = col[i];
    }
}

// 1-hop mean: 4 row-streams in flight, b128 gathers, LDS cross-reduction
__global__ __launch_bounds__(256) void nbr_mean_k(const float* __restrict__ nt,
                                                  const int* __restrict__ row_ptr,
                                                  const int* __restrict__ col_idx,
                                                  float* __restrict__ nbrm) {
    __shared__ int cbuf[256];
    __shared__ v4f red[4][64];
    const int u = blockIdx.x, tid = threadIdx.x;
    const int l6 = tid & 63, rg = tid >> 6;
    const int s = row_ptr[u], e = row_ptr[u + 1];
    v4f acc = {};
    for (int base = s; base < e; base += 256) {
        int nchunk = e - base; if (nchunk > 256) nchunk = 256;
        __syncthreads();
        if (tid < nchunk) cbuf[tid] = col_idx[base + tid];
        __syncthreads();
        for (int j = rg; j < nchunk; j += 4) {
            int c = cbuf[j];
            acc += *(const v4f*)(nt + (size_t)c * HD + l6 * 4);
        }
    }
    red[rg][l6] = acc;
    __syncthreads();
    if (rg == 0) {
        v4f t = red[0][l6] + red[1][l6] + red[2][l6] + red[3][l6];
        int deg = e - s;
        float inv = (deg > 0) ? 1.f / (float)deg : 0.f;
        *(v4f*)(nbrm + (size_t)u * HD + l6 * 4) = t * inv;
    }
}

// edge tokens: 4 edges per block, b128 loads/stores
__global__ __launch_bounds__(256) void edge_tok(const int* __restrict__ row, const int* __restrict__ col,
                                                const float* __restrict__ xWt, const float* __restrict__ xWb,
                                                const float* __restrict__ be, float* __restrict__ out) {
    const int tid = threadIdx.x;
    const int sub = tid >> 6, l6 = tid & 63;
    const int e = blockIdx.x * 4 + sub;
    const int r = row[e], c = col[e];
    v4f t = *(const v4f*)(xWt + (size_t)r * HD + l6 * 4)
          + *(const v4f*)(xWb + (size_t)c * HD + l6 * 4)
          + *(const v4f*)(be + l6 * 4);
    *(v4f*)(out + (size_t)e * HD + l6 * 4) = t;
}

// 2-hop unique-neighbor mean via LDS bitmask + compacted list (exploits sparsity)
__global__ __launch_bounds__(256) void twohop_k(const float* __restrict__ nt,
                                                const int* __restrict__ row_ptr,
                                                const int* __restrict__ col_idx,
                                                float* __restrict__ feat2) {
    __shared__ unsigned bm[NWORDS];
    __shared__ int list[NN];
    __shared__ int lcount;
    __shared__ int nbuf[256];
    __shared__ v4f red[4][64];
    const int u = blockIdx.x, tid = threadIdx.x;
    const int l6 = tid & 63, rg = tid >> 6;
    for (int i = tid; i < NWORDS; i += 256) bm[i] = 0u;
    if (tid == 0) lcount = 0;
    __syncthreads();

    const int s = row_ptr[u], e = row_ptr[u + 1];
    for (int base = s; base < e; base += 256) {
        int nchunk = e - base; if (nchunk > 256) nchunk = 256;
        __syncthreads();
        if (tid < nchunk) nbuf[tid] = col_idx[base + tid];
        __syncthreads();
        // 4 neighbor lists expanded in parallel by 64-lane groups
        for (int j = rg; j < nchunk; j += 4) {
            int v = nbuf[j];
            int vs = row_ptr[v], ve = row_ptr[v + 1];
            for (int p = vs + l6; p < ve; p += 64) {
                int w = col_idx[p];
                atomicOr(&bm[w >> 5], 1u << (w & 31));
            }
        }
    }
    __syncthreads();
    if (tid == 0) atomicAnd(&bm[u >> 5], ~(1u << (u & 31)));   // remove self
    __syncthreads();

    for (int d = tid; d < NWORDS; d += 256) {
        unsigned wrd = bm[d];
        int c = __popc(wrd);
        if (c) {
            int b2 = atomicAdd(&lcount, c);
            while (wrd) {
                int b = __ffs(wrd) - 1;
                list[b2++] = d * 32 + b;
                wrd &= wrd - 1;
            }
        }
    }
    __syncthreads();
    const int cnt = lcount;
    v4f acc = {};
    for (int i = rg; i < cnt; i += 4) {
        int w = list[i];
        acc += *(const v4f*)(nt + (size_t)w * HD + l6 * 4);
    }
    red[rg][l6] = acc;
    __syncthreads();
    if (rg == 0) {
        v4f t = red[0][l6] + red[1][l6] + red[2][l6] + red[3][l6];
        float inv = (cnt > 0) ? 1.f / (float)cnt : 0.f;
        *(v4f*)(feat2 + (size_t)u * HD + l6 * 4) = t * inv;
    }
}

extern "C" void kernel_launch(void* const* d_in, const int* in_sizes, int n_in,
                              void* d_out, int out_size, void* d_ws, size_t ws_size,
                              hipStream_t stream) {
    const float* x   = (const float*)d_in[0];
    const int*   ei  = (const int*)  d_in[1];
    const float* Wn  = (const float*)d_in[2];
    const float* bn  = (const float*)d_in[3];
    const float* We  = (const float*)d_in[4];
    const float* be  = (const float*)d_in[5];
    const float* Ws1 = (const float*)d_in[6];
    const float* bs1 = (const float*)d_in[7];
    const float* Ws2 = (const float*)d_in[8];
    const float* bs2 = (const float*)d_in[9];
    const float* Wh1 = (const float*)d_in[10];
    const float* bh1 = (const float*)d_in[11];
    const float* Wh2 = (const float*)d_in[12];
    const float* bh2 = (const float*)d_in[13];

    const int* row = ei;
    const int* col = ei + NE;

    const size_t NH = (size_t)NN * HD;
    float* out    = (float*)d_out;
    float* nt_out = out;                       // [N,H]
    float* et_out = out + NH;                  // [E,H]
    float* sg_out = et_out + (size_t)NE * HD;  // [N,H]
    float* nh_out = sg_out + NH;               // [N,H]

    // packed f16 weights first (32B-aligned regions), then f32 scratch, then ints
    _Float16* pw   = (_Float16*)d_ws;
    const size_t KSW = 8192;                   // halves per kstep of packing (16 tiles * 32 lanes * 16)
    _Float16* pWn  = pw;                       //  8 ksteps
    _Float16* pWe  = pWn  +  8 * KSW;          // 16 ksteps
    _Float16* pWs1 = pWe  + 16 * KSW;          // 16 ksteps
    _Float16* pWs2 = pWs1 + 16 * KSW;          //  8 ksteps
    _Float16* pWh1 = pWs2 +  8 * KSW;          // 24 ksteps
    _Float16* pWh2 = pWh1 + 24 * KSW;          //  8 ksteps
    float* ws    = (float*)(pWh2 + 8 * KSW);
    float* xWt   = ws;
    float* xWb   = ws + NH;
    float* nbrm  = ws + 2 * NH;
    float* feat2 = ws + 3 * NH;
    float* t1    = ws + 4 * NH;
    int*   iws      = (int*)(ws + 5 * NH);
    int*   row_ptr  = iws;                    // N+1
    int*   cursor   = iws + (NN + 1);         // N
    int*   col_idx  = iws + (2 * NN + 1);     // E
    int*   cnt      = iws + (2 * NN + 1 + NE);// N

    const dim3 blk(256);
    const dim3 gemm_grid(NN / 16);

    // one-time weight repack to fragment-major f16
    pack_w<<<dim3( 8 * 16), dim3(32), 0, stream>>>(Wn,  pWn);
    pack_w<<<dim3(16 * 16), dim3(32), 0, stream>>>(We,  pWe);
    pack_w<<<dim3(16 * 16), dim3(32), 0, stream>>>(Ws1, pWs1);
    pack_w<<<dim3( 8 * 16), dim3(32), 0, stream>>>(Ws2, pWs2);
    pack_w<<<dim3(24 * 16), dim3(32), 0, stream>>>(Wh1, pWh1);
    pack_w<<<dim3( 8 * 16), dim3(32), 0, stream>>>(Wh2, pWh2);

    // node tokens + edge-token factorized GEMMs
    gemm_wmma<<<gemm_grid, blk, 0, stream>>>(x, nullptr, nullptr, pWn, bn, nt_out, NN, 1, 0);
    gemm_wmma<<<gemm_grid, blk, 0, stream>>>(x, nullptr, nullptr, pWe,           nullptr, xWt, NN, 1, 0);
    gemm_wmma<<<gemm_grid, blk, 0, stream>>>(x, nullptr, nullptr, pWe + 8 * KSW, nullptr, xWb, NN, 1, 0);

    // CSR build
    hipMemsetAsync(cnt, 0, NN * sizeof(int), stream);
    count_deg<<<dim3((NE + 255) / 256), blk, 0, stream>>>(row, cnt, NE);
    scan_k<<<dim3(1), blk, 0, stream>>>(cnt, row_ptr, cursor, NN, NE);
    scatter_edges<<<dim3((NE + 255) / 256), blk, 0, stream>>>(row, col, cursor, col_idx, NE);

    // aggregations
    nbr_mean_k<<<dim3(NN), blk, 0, stream>>>(nt_out, row_ptr, col_idx, nbrm);
    edge_tok<<<dim3(NE / 4), blk, 0, stream>>>(row, col, xWt, xWb, be, et_out);
    twohop_k<<<dim3(NN), blk, 0, stream>>>(nt_out, row_ptr, col_idx, feat2);

    // subgraph MLP
    gemm_wmma<<<gemm_grid, blk, 0, stream>>>(nt_out, nbrm, nullptr, pWs1, bs1, t1, NN, 2, 1);
    gemm_wmma<<<gemm_grid, blk, 0, stream>>>(t1, nullptr, nullptr, pWs2, bs2, sg_out, NN, 1, 0);

    // neighborhood MLP
    gemm_wmma<<<gemm_grid, blk, 0, stream>>>(nt_out, nbrm, feat2, pWh1, bh1, t1, NN, 3, 1);
    gemm_wmma<<<gemm_grid, blk, 0, stream>>>(t1, nullptr, nullptr, pWh2, bh2, nh_out, NN, 1, 0);
}
